// ABSATokenAggregator_8615704396313
// MI455X (gfx1250) — matrix operand order repaired and checked by reading, not compiled, not run on hardware
//
#include <hip/hip_runtime.h>
#include <hip/hip_fp16.h>

typedef __attribute__((ext_vector_type(16))) _Float16 v16h;
typedef __attribute__((ext_vector_type(8)))  _Float16 v8h;
typedef __attribute__((ext_vector_type(4)))  _Float16 v4h;
typedef __attribute__((ext_vector_type(8)))  float    v8f;
typedef __attribute__((ext_vector_type(4)))  unsigned int u32x4;
typedef __attribute__((ext_vector_type(8)))  int      i32x8;
typedef __attribute__((ext_vector_type(4)))  int      i32x4;

#define B_   64
#define S_   512
#define D_   1024
#define W_   256
#define A_   8
#define NSEG (B_ * W_)   // 16384
#define NTOK (B_ * S_)   // 32768

// ---------------------------------------------------------------------------
// 1) enh_w (f32, row-major NxK) -> f16 copy (rows of enh_w are exactly the
//    contiguous K-runs the WMMA B-fragment wants, so no transpose needed).
// ---------------------------------------------------------------------------
__global__ void k_convert_w(const float* __restrict__ w, _Float16* __restrict__ o) {
    int i = (blockIdx.x * blockDim.x + threadIdx.x) * 4;
    float4 v = *(const float4*)(w + i);
    v4h h;
    h.x = (_Float16)v.x; h.y = (_Float16)v.y; h.z = (_Float16)v.z; h.w = (_Float16)v.w;
    *(v4h*)(o + i) = h;
}

// ---------------------------------------------------------------------------
// 2) per-token attention score: scores[t] = dot(bert[t,:], attn_w) + attn_b
// ---------------------------------------------------------------------------
__global__ void k_scores(const float* __restrict__ X, const float* __restrict__ aw,
                         const float* __restrict__ ab, float* __restrict__ sc) {
    int wave = (blockIdx.x * blockDim.x + threadIdx.x) >> 5;
    int lane = threadIdx.x & 31;
    const float4* xr = (const float4*)(X + (size_t)wave * D_);
    const float4* wr = (const float4*)aw;
    float acc = 0.f;
#pragma unroll
    for (int i = 0; i < 8; ++i) {
        float4 a = xr[lane + i * 32];
        float4 b = wr[lane + i * 32];
        acc += a.x * b.x + a.y * b.y + a.z * b.z + a.w * b.w;
    }
#pragma unroll
    for (int m = 16; m >= 1; m >>= 1) acc += __shfl_xor(acc, m, 32);
    if (lane == 0) sc[wave] = acc + ab[0];
}

// ---------------------------------------------------------------------------
// 3) word_ids are SORTED per row -> segments are contiguous token runs.
// ---------------------------------------------------------------------------
__global__ void k_segments(const int* __restrict__ wid, const float* __restrict__ sc,
                           int* __restrict__ seg_cnt, int* __restrict__ seg_start,
                           float* __restrict__ seg_denom, float* __restrict__ expw) {
    __shared__ int   sseg[S_];
    __shared__ float ssc[S_];
    int b = blockIdx.x, t = threadIdx.x;
    if (t < W_) seg_cnt[b * W_ + t] = 0;           // empty segments -> cnt 0
    int w = wid[b * S_ + t];
    sseg[t] = w;
    ssc[t]  = sc[b * S_ + t];
    __syncthreads();
    bool head = (t == 0) || (sseg[t - 1] != w);
    if (head) {
        int e = t + 1;
        while (e < S_ && sseg[e] == w) ++e;
        float mx = ssc[t];
        for (int i = t + 1; i < e; ++i) mx = fmaxf(mx, ssc[i]);
        float dn = 0.f;
        for (int i = t; i < e; ++i) {
            float ev = __expf(ssc[i] - mx);
            expw[b * S_ + i] = ev;
            dn += ev;
        }
        int sg = b * W_ + w;
        seg_cnt[sg]   = e - t;
        seg_start[sg] = b * S_ + t;
        seg_denom[sg] = dn;
    }
}

// ---------------------------------------------------------------------------
// 4) per-segment weighted sum / denom, gate, + pos_emb, store sent as f16.
//    Second read of bert_output hits the 192 MB L2 (bert is 128 MB).
// ---------------------------------------------------------------------------
__global__ void k_aggregate(const float* __restrict__ X, const float* __restrict__ gw,
                            const float* __restrict__ gb, const float* __restrict__ pos,
                            const int* __restrict__ seg_cnt, const int* __restrict__ seg_start,
                            const float* __restrict__ seg_denom, const float* __restrict__ expw,
                            _Float16* __restrict__ senth) {
    __shared__ float red[8];
    int sg = blockIdx.x, t = threadIdx.x;
    int d  = t * 4;
    int cnt = seg_cnt[sg];
    float ax = 0.f, ay = 0.f, az = 0.f, aw_ = 0.f;
    if (cnt > 0) {
        int st = seg_start[sg];
        for (int i = 0; i < cnt; ++i) {
            float ev = expw[st + i];
            float4 x = *(const float4*)(X + (size_t)(st + i) * D_ + d);
            ax += x.x * ev; ay += x.y * ev; az += x.z * ev; aw_ += x.w * ev;
        }
        float inv = 1.f / fmaxf(seg_denom[sg], 1e-30f);
        ax *= inv; ay *= inv; az *= inv; aw_ *= inv;
        float4 g4 = *(const float4*)(gw + d);
        float p = ax * g4.x + ay * g4.y + az * g4.z + aw_ * g4.w;
#pragma unroll
        for (int m = 16; m >= 1; m >>= 1) p += __shfl_xor(p, m, 32);
        if ((t & 31) == 0) red[t >> 5] = p;
        __syncthreads();
        float tot = red[0] + red[1] + red[2] + red[3] + red[4] + red[5] + red[6] + red[7];
        float g = 1.f / (1.f + __expf(-(tot + gb[0])));
        float4 pe = *(const float4*)(pos + (size_t)(sg % W_) * D_ + d);
        ax = ax * g + pe.x; ay = ay * g + pe.y; az = az * g + pe.z; aw_ = aw_ * g + pe.w;
    }
    v4h h;
    h.x = (_Float16)ax; h.y = (_Float16)ay; h.z = (_Float16)az; h.w = (_Float16)aw_;
    *(v4h*)(senth + (size_t)sg * D_ + d) = h;
}

// ---------------------------------------------------------------------------
// 5) stable-argsort of 0/1 keys == indices of zeros in ascending order.
// ---------------------------------------------------------------------------
__global__ void k_gather(const int* __restrict__ araw, const _Float16* __restrict__ senth,
                         _Float16* __restrict__ asph, float* __restrict__ wl) {
    __shared__ int sidx[A_];
    __shared__ int sminc;
    int b = blockIdx.x, t = threadIdx.x;
    if (t == 0) {
        int c = 0;
        for (int w = 0; w < W_; ++w)
            if (araw[b * W_ + w] == 0) { if (c < A_) sidx[c] = w; ++c; }
        int minc = c < A_ ? c : A_;
        sminc = minc;
        for (int a = 0; a < A_; ++a) wl[b * A_ + a] = (a < minc) ? 1.f : 0.f;
    }
    __syncthreads();
    int minc = sminc;
    int d = t * 4;
#pragma unroll
    for (int a = 0; a < A_; ++a) {
        v4h h = {};
        if (a < minc)
            h = *(const v4h*)(senth + (size_t)(b * W_ + sidx[a]) * D_ + d);
        *(v4h*)(asph + (size_t)(b * A_ + a) * D_ + d) = h;
    }
}

// ---------------------------------------------------------------------------
// 6) WMMA f16 GEMM with TDM double buffering.
//    Tensor DMA descriptor (CDNA5 ISA):
//      group0: count=1 | lds_addr | global_addr | type=2
//      group1: data_size=2B, pad_enable (16 B pad after every 64 B row ->
//              LDS row stride 80 B = 40 halfs, matching LDSP), tile 32x128,
//              tensor_dim0=1024, tensor_dim0_stride=1024, tensor_dim1=Mrows.
//    Wave 0 issues DMA for K-step k+1 into back buffer, s_wait_tensorcnt 2
//    (TDM in-order per wave => front buffer's 2 ops done), barrier, WMMA.
// ---------------------------------------------------------------------------
#define BM 128
#define BN 128
#define BK 32
#define LDSP (BK + 8)   // 40 halfs = 80 B stride (created by TDM pad)

__device__ __forceinline__ void tdm_load_tile(const _Float16* gptr, void* lds,
                                              int rows_total) {
    unsigned long long ga = (unsigned long long)(uintptr_t)gptr;
    u32x4 g0;
    g0.x = 1u;                                   // count=1, user mode
    g0.y = (unsigned int)(uintptr_t)lds;         // LDS byte address (low 32 bits)
    g0.z = (unsigned int)ga;                     // global_addr[31:0]
    g0.w = ((unsigned int)(ga >> 32) & 0x1FFFFFFu) | (2u << 30); // [56:32] | type=2
    i32x8 g1;
    g1[0] = (1 << 16) | (1 << 20) | (3 << 22) | (3 << 25);
    //       data_size=2B  pad_en   interval=64B  amount=16B
    g1[1] = (1024 << 16);                        // tensor_dim0 = 1024 (bits 63:48)
    g1[2] = (rows_total & 0xFFFF) << 16;         // tensor_dim1 lo16 (bits 95:80)
    g1[3] = (32 << 16);                          // tile_dim0 = 32 (bits 127:112)
    g1[4] = 128;                                 // tile_dim1 = 128 (bits 143:128)
    g1[5] = 1024;                                // tensor_dim0_stride lo32
    g1[6] = 0;
    g1[7] = 0;
    i32x4 z4 = {0, 0, 0, 0};                     // 2D: groups 2/3 unused
#if defined(__clang_major__) && (__clang_major__ >= 23)
    i32x8 z8 = {0, 0, 0, 0, 0, 0, 0, 0};
    __builtin_amdgcn_tensor_load_to_lds(g0, g1, z4, z4, z8, 0);
#else
    __builtin_amdgcn_tensor_load_to_lds(g0, g1, z4, z4, 0);
#endif
}

__global__ __launch_bounds__(256)
void k_gemm_bias(const _Float16* __restrict__ Xh, const _Float16* __restrict__ Wh,
                 const float* __restrict__ bias, float* __restrict__ out, int Mrows) {
    __shared__ _Float16 At[2][BM * LDSP];
    __shared__ _Float16 Bt[2][BN * LDSP];

    const int K    = D_;
    const int tid  = threadIdx.x;
    const int lane = tid & 31;
    const int wid  = tid >> 5;
    const int lh   = lane >> 4;    // lane-group (0/1)
    const int ln   = lane & 15;
    const int wm   = wid & 1;      // 2 wave-rows  (64 M each)
    const int wn   = wid >> 1;     // 4 wave-cols  (32 N each)
    const int m0   = blockIdx.x * BM;
    const int n0   = blockIdx.y * BN;

    const v8f zero = {0.f, 0.f, 0.f, 0.f, 0.f, 0.f, 0.f, 0.f};
    v8f acc[4][2];
#pragma unroll
    for (int mt = 0; mt < 4; ++mt)
#pragma unroll
        for (int nt = 0; nt < 2; ++nt) acc[mt][nt] = zero;

    // prologue: wave 0 kicks DMA for K-step 0 into buffer 0
    if (tid < 32) {
        tdm_load_tile(Xh + (size_t)m0 * K, At[0], Mrows);
        tdm_load_tile(Wh + (size_t)n0 * K, Bt[0], D_);
    }

    for (int kt = 0; kt < K; kt += BK) {
        const int cur = (kt >> 5) & 1;
        if (tid < 32) {
            if (kt + BK < K) {
                tdm_load_tile(Xh + (size_t)m0 * K + kt + BK, At[cur ^ 1], Mrows);
                tdm_load_tile(Wh + (size_t)n0 * K + kt + BK, Bt[cur ^ 1], D_);
                __builtin_amdgcn_s_wait_tensorcnt(2);  // front buffer complete
            } else {
                __builtin_amdgcn_s_wait_tensorcnt(0);
            }
        }
        __syncthreads();   // publish front buffer to all waves

        const _Float16* Ac = At[cur];
        const _Float16* Bc = Bt[cur];

        // B fragments: lane = column N, K = 16*lh + e (contiguous run)
        v16h bf[2];
#pragma unroll
        for (int nt = 0; nt < 2; ++nt) {
            const _Float16* p = Bc + (wn * 32 + nt * 16 + ln) * LDSP + 16 * lh;
            v8h x0 = *(const v8h*)(p);
            v8h x1 = *(const v8h*)(p + 8);
            bf[nt] = __builtin_shufflevector(x0, x1, 0, 1, 2, 3, 4, 5, 6, 7,
                                                     8, 9, 10, 11, 12, 13, 14, 15);
        }
        // A fragments: lane = row M, K = 16*lh + e (contiguous run)
#pragma unroll
        for (int mt = 0; mt < 4; ++mt) {
            const _Float16* p = Ac + (wm * 64 + mt * 16 + ln) * LDSP;
            v8h x0 = *(const v8h*)(p + 16 * lh);
            v8h x1 = *(const v8h*)(p + 16 * lh + 8);
            v16h af = __builtin_shufflevector(x0, x1, 0, 1, 2, 3, 4, 5, 6, 7,
                                                      8, 9, 10, 11, 12, 13, 14, 15);
#pragma unroll
            for (int nt = 0; nt < 2; ++nt)
                acc[mt][nt] = __builtin_amdgcn_wmma_f32_16x16x32_f16(
                    false, af, false, bf[nt], (short)0, acc[mt][nt], false, false);
        }
        __syncthreads();   // all readers done before wave 0 reuses this buffer
    }

    // epilogue: D layout M = r + 8*lh, N = ln ; fuse bias add
#pragma unroll
    for (int mt = 0; mt < 4; ++mt) {
#pragma unroll
        for (int nt = 0; nt < 2; ++nt) {
            int n = n0 + wn * 32 + nt * 16 + ln;
            float bb = bias[n];
#pragma unroll
            for (int r = 0; r < 8; ++r) {
                int m = m0 + wm * 64 + mt * 16 + r + 8 * lh;
                out[(size_t)m * D_ + n] = acc[mt][nt][r] + bb;
            }
        }
    }
}

// ---------------------------------------------------------------------------
// 7) in-place LayerNorm * ln_g + ln_b, then ReLU (out1+out2 contiguous rows).
// ---------------------------------------------------------------------------
__global__ void k_ln_relu(float* __restrict__ H, const float* __restrict__ g,
                          const float* __restrict__ bvec) {
    __shared__ float rs[8], rq[8];
    int row = blockIdx.x, t = threadIdx.x;
    float* p = H + (size_t)row * D_;
    float4 h = *(const float4*)(p + t * 4);
    float s = h.x + h.y + h.z + h.w;
    float q = h.x * h.x + h.y * h.y + h.z * h.z + h.w * h.w;
#pragma unroll
    for (int m = 16; m >= 1; m >>= 1) {
        s += __shfl_xor(s, m, 32);
        q += __shfl_xor(q, m, 32);
    }
    if ((t & 31) == 0) { rs[t >> 5] = s; rq[t >> 5] = q; }
    __syncthreads();
    float S = 0.f, Q = 0.f;
#pragma unroll
    for (int i = 0; i < 8; ++i) { S += rs[i]; Q += rq[i]; }
    float mu  = S * (1.f / D_);
    float var = Q * (1.f / D_) - mu * mu;
    float inv = rsqrtf(var + 1e-5f);
    float4 gg = *(const float4*)(g + t * 4);
    float4 bb = *(const float4*)(bvec + t * 4);
    h.x = fmaxf((h.x - mu) * inv * gg.x + bb.x, 0.f);
    h.y = fmaxf((h.y - mu) * inv * gg.y + bb.y, 0.f);
    h.z = fmaxf((h.z - mu) * inv * gg.z + bb.z, 0.f);
    h.w = fmaxf((h.w - mu) * inv * gg.w + bb.w, 0.f);
    *(float4*)(p + t * 4) = h;
}

// ---------------------------------------------------------------------------
extern "C" void kernel_launch(void* const* d_in, const int* in_sizes, int n_in,
                              void* d_out, int out_size, void* d_ws, size_t ws_size,
                              hipStream_t stream) {
    (void)in_sizes; (void)n_in; (void)out_size; (void)ws_size;
    const float* bert   = (const float*)d_in[0];
    const int*   wid    = (const int*)  d_in[1];
    const int*   araw   = (const int*)  d_in[2];
    const float* attn_w = (const float*)d_in[3];
    const float* attn_b = (const float*)d_in[4];
    const float* gate_w = (const float*)d_in[5];
    const float* gate_b = (const float*)d_in[6];
    const float* pos    = (const float*)d_in[7];
    const float* enh_w  = (const float*)d_in[8];
    const float* enh_b  = (const float*)d_in[9];
    const float* ln_g   = (const float*)d_in[10];
    const float* ln_b   = (const float*)d_in[11];

    char* wsp = (char*)d_ws;
    auto alloc = [&](size_t bytes) -> char* {
        char* p = wsp;
        wsp += (bytes + 255) & ~(size_t)255;
        return p;
    };
    _Float16* Wh        = (_Float16*)alloc((size_t)D_ * D_ * 2);       // 2 MB
    float*    scores    = (float*)   alloc((size_t)NTOK * 4);
    float*    expw      = (float*)   alloc((size_t)NTOK * 4);
    int*      seg_cnt   = (int*)     alloc((size_t)NSEG * 4);
    int*      seg_start = (int*)     alloc((size_t)NSEG * 4);
    float*    seg_denom = (float*)   alloc((size_t)NSEG * 4);
    _Float16* senth     = (_Float16*)alloc((size_t)NSEG * D_ * 2);     // 32 MB
    _Float16* asph      = (_Float16*)alloc((size_t)B_ * A_ * D_ * 2);  // 1 MB

    float* out1 = (float*)d_out;                       // (B, W, D)
    float* out2 = out1 + (size_t)NSEG * D_;            // (B, A, D)
    float* wl   = out2 + (size_t)B_ * A_ * D_;         // (B, A) as float

    k_convert_w<<<D_ * D_ / (256 * 4), 256, 0, stream>>>(enh_w, Wh);
    k_scores   <<<NTOK / 8, 256, 0, stream>>>(bert, attn_w, attn_b, scores);
    k_segments <<<B_, S_, 0, stream>>>(wid, scores, seg_cnt, seg_start, seg_denom, expw);
    k_aggregate<<<NSEG, 256, 0, stream>>>(bert, gate_w, gate_b, pos,
                                          seg_cnt, seg_start, seg_denom, expw, senth);
    k_gather   <<<B_, 256, 0, stream>>>(araw, senth, asph, wl);

    dim3 g1(NSEG / BM, D_ / BN);
    k_gemm_bias<<<g1, 256, 0, stream>>>(senth, Wh, enh_b, out1, NSEG);
    dim3 g2((B_ * A_) / BM, D_ / BN);
    k_gemm_bias<<<g2, 256, 0, stream>>>(asph, Wh, enh_b, out2, B_ * A_);

    k_ln_relu<<<NSEG + B_ * A_, 256, 0, stream>>>(out1, ln_g, ln_b);
}
